// SpecGNO_58643483459742
// MI455X (gfx1250) — compile-verified, matrix operand order checked by reading, not run on hardware
//
#include <hip/hip_runtime.h>
#include <cstdint>
#include <cstddef>

// Static problem dims (match reference)
#define T_DIM   8
#define N_NODES 8000
#define E_EDGES 32000
#define C_CH    32
#define F_FEAT  3
#define MODESN  4
#define TDIMN   32
#define KWN     64
#define NC      (N_NODES * C_CH)      // 256000
#define CC      (C_CH * C_CH)         // 1024
#define ETPW    8                     // edge tiles per wave in the WMMA GEMM

typedef float     v2f  __attribute__((ext_vector_type(2)));
typedef float     v8f  __attribute__((ext_vector_type(8)));
typedef _Float16  v16h __attribute__((ext_vector_type(16)));

// 8-point DFT twiddle tables: cos(2*pi*p/8), sin(2*pi*p/8)
__device__ __constant__ float C8[8] = { 1.f,  0.70710678f, 0.f, -0.70710678f, -1.f, -0.70710678f,  0.f,  0.70710678f};
__device__ __constant__ float S8[8] = { 0.f,  0.70710678f, 1.f,  0.70710678f,  0.f, -0.70710678f, -1.f, -0.70710678f};

// ---------------- time-embedding projection: tproj[t][c] = te[t] @ emb_w[3:,:] + emb_b ----
__global__ void time_proj_kernel(const float* __restrict__ emb_w,
                                 const float* __restrict__ emb_b,
                                 float* __restrict__ tproj) {
    int tid = threadIdx.x;            // 256 threads = (T=8) x (C=32)
    int t = tid >> 5, c = tid & 31;
    float acc = emb_b[c];
    for (int d = 0; d < TDIMN; ++d) {
        int j = d & 15;               // half = 16, freqs[j] = exp(-ln(1e4)*j/(half-1))
        float fr = expf(-logf(10000.f) * (float)j / 15.f);
        float a  = (float)t * fr;
        float te = (d < 16) ? sinf(a) : cosf(a);
        acc += te * emb_w[(F_FEAT + d) * C_CH + c];
    }
    tproj[t * C_CH + c] = acc;
}

// ---------------- h[t,n,c] = x[n,:] @ emb_w[:3,c] + tproj[t,c] ----------------------------
__global__ void embed_kernel(const float* __restrict__ x,
                             const float* __restrict__ emb_w,
                             const float* __restrict__ tproj,
                             float* __restrict__ h) {
    int idx = blockIdx.x * blockDim.x + threadIdx.x;   // over N*C
    if (idx >= NC) return;
    int n = idx >> 5, c = idx & 31;
    float base = x[n * 3 + 0] * emb_w[0 * C_CH + c]
               + x[n * 3 + 1] * emb_w[1 * C_CH + c]
               + x[n * 3 + 2] * emb_w[2 * C_CH + c];
    #pragma unroll
    for (int t = 0; t < T_DIM; ++t)
        h[t * NC + idx] = base + tproj[t * C_CH + c];
}

// ---------------- edge MLP layers 0,1: K[e,0:64] = relu(relu(ea@W0+b0)@W1+b1) -------------
__global__ __launch_bounds__(256) void edge_k01_kernel(const float* __restrict__ ea,
                                                       const float* __restrict__ w0, const float* __restrict__ b0,
                                                       const float* __restrict__ w1, const float* __restrict__ b1,
                                                       float* __restrict__ Kbuf) {
    __shared__ float s0[8][KWN];
    int lane = threadIdx.x & 31, wv = threadIdx.x >> 5;
    int e = blockIdx.x * 8 + wv;                        // one wave per edge
    float a[6];
    #pragma unroll
    for (int f = 0; f < 6; ++f) a[f] = ea[e * 6 + f];
    #pragma unroll
    for (int half = 0; half < 2; ++half) {
        int u = lane + half * 32;
        float acc = b0[u];
        #pragma unroll
        for (int f = 0; f < 6; ++f) acc += a[f] * w0[f * KWN + u];
        s0[wv][u] = fmaxf(acc, 0.f);
    }
    __syncthreads();
    #pragma unroll
    for (int half = 0; half < 2; ++half) {
        int u = lane + half * 32;
        float acc = b1[u];
        #pragma unroll
        for (int j = 0; j < KWN; ++j) acc += s0[wv][j] * w1[j * KWN + u];
        Kbuf[e * KWN + u] = fmaxf(acc, 0.f);
    }
}

// ---------------- edge MLP layer 2 (dominant GEMM) via WMMA -------------------------------
// W_edge(16e x 16col tile) = K(16x64) @ k2_w(64x1024) + b2.
// Each wave owns one 16-col tile: B fragments (16 x v2f = 32 VGPRs) are hoisted into
// registers once and reused across ETPW edge tiles -> steady-state loop is
// 16 x (global_load_b64 A + v_wmma_f32_16x16x4_f32) + 8 stores.
#if __has_builtin(__builtin_amdgcn_wmma_f32_16x16x4_f32)
__global__ __launch_bounds__(128) void k2_gemm_wmma(const float* __restrict__ Kbuf,
                                                    const float* __restrict__ w2,
                                                    const float* __restrict__ b2,
                                                    float* __restrict__ Wedge) {
    const int lane = threadIdx.x & 31;
    const int wv   = threadIdx.x >> 5;
    const int ct   = blockIdx.x * 4 + wv;        // col tile 0..63
    const int et0  = blockIdx.y * ETPW;          // first edge tile for this wave
    const int m    = lane & 15;                  // A: row M; B/C/D: col N
    const int kh   = lane >> 4;                  // K-pair half select
    const int col  = ct * 16 + m;

    // hoist B fragments (one 16-col tile of k2_w) into registers
    v2f bfrag[16];
    #pragma unroll
    for (int kk = 0; kk < 16; ++kk) {
        int kbase = kk * 4 + kh * 2;
        bfrag[kk].x = w2[(kbase + 0) * CC + col];
        bfrag[kk].y = w2[(kbase + 1) * CC + col];
    }
    float bias = b2[col];                        // per-column bias, replicated over M rows

    for (int ei = 0; ei < ETPW; ++ei) {
        const int et = et0 + ei;
        const int e  = et * 16 + m;
        v8f cacc;
        #pragma unroll
        for (int i = 0; i < 8; ++i) cacc[i] = bias;
        #pragma unroll
        for (int kk = 0; kk < 16; ++kk) {        // K = 64 in steps of 4
            int kbase = kk * 4 + kh * 2;
            v2f afrag = *(const v2f*)(Kbuf + e * KWN + kbase);   // A[m][kbase..kbase+1]
            cacc = __builtin_amdgcn_wmma_f32_16x16x4_f32(false, afrag, false, bfrag[kk],
                                                         (short)0, cacc, false, false);
        }
        #pragma unroll
        for (int v = 0; v < 8; ++v) {            // D VGPR v -> M = v + 8*kh
            int er = et * 16 + v + 8 * kh;
            Wedge[(size_t)er * CC + ct * 16 + (lane & 15)] = cacc[v];
        }
    }
}
#else
// Fallback: f16 inputs, f32 accumulate (codegen-verified builtin). K=64 -> two 16x16x32 WMMAs.
__global__ __launch_bounds__(128) void k2_gemm_wmma(const float* __restrict__ Kbuf,
                                                    const float* __restrict__ w2,
                                                    const float* __restrict__ b2,
                                                    float* __restrict__ Wedge) {
    const int lane = threadIdx.x & 31;
    const int wv   = threadIdx.x >> 5;
    const int ct   = blockIdx.x * 4 + wv;
    const int et0  = blockIdx.y * ETPW;
    const int m    = lane & 15;
    const int kh   = lane >> 4;
    const int col  = ct * 16 + m;

    v16h bfrag[2];
    #pragma unroll
    for (int kb = 0; kb < KWN; kb += 32) {
        #pragma unroll
        for (int i = 0; i < 8; ++i) {
            bfrag[kb / 32][i]     = (_Float16)w2[(kb + kh * 8 + i) * CC + col];
            bfrag[kb / 32][8 + i] = (_Float16)w2[(kb + 16 + kh * 8 + i) * CC + col];
        }
    }
    float bias = b2[col];

    for (int ei = 0; ei < ETPW; ++ei) {
        const int et = et0 + ei;
        const int e  = et * 16 + m;
        v8f cacc;
        #pragma unroll
        for (int i = 0; i < 8; ++i) cacc[i] = bias;
        #pragma unroll
        for (int kb = 0; kb < KWN; kb += 32) {
            v16h afrag;
            #pragma unroll
            for (int i = 0; i < 8; ++i) {
                afrag[i]     = (_Float16)Kbuf[e * KWN + kb + kh * 8 + i];
                afrag[8 + i] = (_Float16)Kbuf[e * KWN + kb + 16 + kh * 8 + i];
            }
            cacc = __builtin_amdgcn_wmma_f32_16x16x32_f16(false, afrag, false, bfrag[kb / 32],
                                                          (short)0, cacc, false, false);
        }
        #pragma unroll
        for (int v = 0; v < 8; ++v) {
            int er = et * 16 + v + 8 * kh;
            Wedge[(size_t)er * CC + ct * 16 + (lane & 15)] = cacc[v];
        }
    }
}
#endif

// ---------------- misc helpers ------------------------------------------------------------
__global__ void zero_f(float* __restrict__ p, int n) {
    int i = blockIdx.x * blockDim.x + threadIdx.x;
    if (i < n) p[i] = 0.f;
}

__global__ void count_kernel(const int* __restrict__ ei, float* __restrict__ cnt) {
    int e = blockIdx.x * blockDim.x + threadIdx.x;
    if (e < E_EDGES) atomicAdd(&cnt[ei[E_EDGES + e]], 1.f);
}

// ---------------- rfft over T=8 (modes 0..3; mode 4 zeroed by reference) ------------------
__global__ void fft_kernel(const float* __restrict__ h,
                           float* __restrict__ xfr, float* __restrict__ xfi) {
    int idx = blockIdx.x * blockDim.x + threadIdx.x;  // over N*C
    if (idx >= NC) return;
    float v[T_DIM];
    #pragma unroll
    for (int t = 0; t < T_DIM; ++t) v[t] = h[t * NC + idx];
    #pragma unroll
    for (int k = 0; k < MODESN; ++k) {
        float xr = 0.f, xi = 0.f;
        #pragma unroll
        for (int t = 0; t < T_DIM; ++t) {
            int p = (k * t) & 7;
            xr += v[t] * C8[p];
            xi -= v[t] * S8[p];       // exp(-i*theta)
        }
        xfr[k * NC + idx] = xr;
        xfi[k * NC + idx] = xi;
    }
}

// ---------------- spectral channel-mix + irfft + SiLU, one wave per node ------------------
// irfft(T=8): x[t] = (X0.re + 2*sum_{k=1..3} (Xk.re*cos - Xk.im*sin)) / 8   (mode4 = 0)
__global__ __launch_bounds__(256) void spec_kernel(const float* __restrict__ xfr,
                                                   const float* __restrict__ xfi,
                                                   const float* __restrict__ wr,
                                                   const float* __restrict__ wi,
                                                   float* __restrict__ h) {
    int lane = threadIdx.x & 31, wv = threadIdx.x >> 5;
    int n = blockIdx.x * 8 + wv;
    float xr[MODESN], xi[MODESN];
    #pragma unroll
    for (int k = 0; k < MODESN; ++k) {
        xr[k] = xfr[k * NC + n * 32 + lane];
        xi[k] = xfi[k * NC + n * 32 + lane];
    }
    float orr[MODESN], oii[MODESN];
    #pragma unroll
    for (int k = 0; k < MODESN; ++k) {
        float ar = 0.f, ai = 0.f;
        #pragma unroll
        for (int c = 0; c < 32; ++c) {     // constant-lane shfl -> wave-uniform broadcast
            float xrc = __shfl(xr[k], c);
            float xic = __shfl(xi[k], c);
            float wrc = wr[(k * 32 + c) * 32 + lane];
            float wic = wi[(k * 32 + c) * 32 + lane];
            ar += xrc * wrc - xic * wic;
            ai += xrc * wic + xic * wrc;
        }
        orr[k] = ar; oii[k] = ai;
    }
    #pragma unroll
    for (int t = 0; t < T_DIM; ++t) {
        float val = orr[0];
        #pragma unroll
        for (int k = 1; k < MODESN; ++k) {
            int p = (k * t) & 7;
            val += 2.f * (orr[k] * C8[p] - oii[k] * S8[p]);
        }
        val *= 0.125f;
        float s = val / (1.f + expf(-val));   // SiLU
        h[t * NC + n * 32 + lane] = s;
    }
}

// ---------------- NNConv messages: one wave per edge, scatter-add into agg ----------------
__global__ __launch_bounds__(256) void message_kernel(const float* __restrict__ h,
                                                      const float* __restrict__ Wedge,
                                                      const int* __restrict__ ei,
                                                      float* __restrict__ agg) {
    int lane = threadIdx.x & 31, wv = threadIdx.x >> 5;
    int e = blockIdx.x * 8 + wv;
    int src = ei[e], tgt = ei[E_EDGES + e];
    float w[32];                              // column `lane` of W_edge[e]
    #pragma unroll
    for (int c = 0; c < 32; ++c) w[c] = Wedge[(size_t)e * CC + c * 32 + lane];
    #pragma unroll
    for (int t = 0; t < T_DIM; ++t) {
        float hv = h[t * NC + src * 32 + lane];
        float acc = 0.f;
        #pragma unroll
        for (int c = 0; c < 32; ++c) acc += __shfl(hv, c) * w[c];
        atomicAdd(&agg[t * NC + tgt * 32 + lane], acc);
    }
}

// ---------------- combine: h = relu(agg/denom + h@root_w + root_b), in place --------------
__global__ __launch_bounds__(256) void combine_kernel(float* __restrict__ h,
                                                      const float* __restrict__ agg,
                                                      const float* __restrict__ cnt,
                                                      const float* __restrict__ root_w,
                                                      const float* __restrict__ root_b) {
    int lane = threadIdx.x & 31, wv = threadIdx.x >> 5;
    int tn = blockIdx.x * 8 + wv;             // t*N + n
    int n  = tn % N_NODES;
    int base = tn * 32;                       // == t*NC + n*32
    float hv = h[base + lane];
    float acc = root_b[lane];
    #pragma unroll
    for (int c = 0; c < 32; ++c) acc += __shfl(hv, c) * root_w[c * 32 + lane];
    float d = fmaxf(cnt[n], 1.f);
    float val = agg[base + lane] / d + acc;
    h[base + lane] = fmaxf(val, 0.f);
}

// ---------------- output projection: out[t,n,:3] = h[t,n,:] @ inv_w + inv_b ---------------
__global__ void out_kernel(const float* __restrict__ h,
                           const float* __restrict__ inv_w,
                           const float* __restrict__ inv_b,
                           float* __restrict__ out) {
    int tn = blockIdx.x * blockDim.x + threadIdx.x;
    if (tn >= T_DIM * N_NODES) return;
    const float* row = h + (size_t)tn * 32;
    float a0 = inv_b[0], a1 = inv_b[1], a2 = inv_b[2];
    #pragma unroll
    for (int c = 0; c < 32; ++c) {
        float v = row[c];
        a0 += v * inv_w[c * 3 + 0];
        a1 += v * inv_w[c * 3 + 1];
        a2 += v * inv_w[c * 3 + 2];
    }
    out[tn * 3 + 0] = a0; out[tn * 3 + 1] = a1; out[tn * 3 + 2] = a2;
}

// ==========================================================================================
extern "C" void kernel_launch(void* const* d_in, const int* in_sizes, int n_in,
                              void* d_out, int out_size, void* d_ws, size_t ws_size,
                              hipStream_t stream) {
    (void)in_sizes; (void)n_in; (void)out_size; (void)ws_size;
    const float* x      = (const float*)d_in[0];
    const float* eattr  = (const float*)d_in[1];
    const float* emb_w  = (const float*)d_in[2];
    const float* emb_b  = (const float*)d_in[3];
    const float* k0_w   = (const float*)d_in[4];
    const float* k0_b   = (const float*)d_in[5];
    const float* k1_w   = (const float*)d_in[6];
    const float* k1_b   = (const float*)d_in[7];
    const float* k2_w   = (const float*)d_in[8];
    const float* k2_b   = (const float*)d_in[9];
    const float* root_w = (const float*)d_in[10];
    const float* root_b = (const float*)d_in[11];
    const float* tc_wr  = (const float*)d_in[12];
    const float* tc_wi  = (const float*)d_in[13];
    const float* inv_w  = (const float*)d_in[14];
    const float* inv_b  = (const float*)d_in[15];
    const int*   ei     = (const int*)d_in[16];
    float* out = (float*)d_out;

    // workspace layout (floats); total ~164 MB
    float* ws    = (float*)d_ws;
    float* h     = ws;                       // T*N*C        = 2,048,000
    float* Kbuf  = h     + (size_t)T_DIM * NC;          // E*KW = 2,048,000
    float* xfr   = Kbuf  + (size_t)E_EDGES * KWN;       // 4*N*C = 1,024,000
    float* xfi   = xfr   + (size_t)MODESN * NC;         // 1,024,000
    float* agg   = xfi   + (size_t)MODESN * NC;         // T*N*C = 2,048,000
    float* cnt   = agg   + (size_t)T_DIM * NC;          // N = 8,000
    float* tproj = cnt   + N_NODES;                     // T*C = 256
    float* Wedge = tproj + T_DIM * C_CH;                // E*C*C = 32,768,000

    // embedding
    time_proj_kernel<<<1, 256, 0, stream>>>(emb_w, emb_b, tproj);
    embed_kernel<<<NC / 256, 256, 0, stream>>>(x, emb_w, tproj, h);

    // edge kernel MLP (shared across timesteps & layers)
    edge_k01_kernel<<<E_EDGES / 8, 256, 0, stream>>>(eattr, k0_w, k0_b, k1_w, k1_b, Kbuf);
    k2_gemm_wmma<<<dim3(16, E_EDGES / (16 * ETPW)), 128, 0, stream>>>(Kbuf, k2_w, k2_b, Wedge);

    // scatter-mean denominators
    zero_f<<<(N_NODES + 255) / 256, 256, 0, stream>>>(cnt, N_NODES);
    count_kernel<<<E_EDGES / 256, 256, 0, stream>>>(ei, cnt);

    for (int layer = 0; layer < 2; ++layer) {
        fft_kernel<<<NC / 256, 256, 0, stream>>>(h, xfr, xfi);
        spec_kernel<<<N_NODES / 8, 256, 0, stream>>>(xfr, xfi,
                                                     tc_wr + layer * MODESN * CC,
                                                     tc_wi + layer * MODESN * CC, h);
        zero_f<<<(T_DIM * NC) / 256, 256, 0, stream>>>(agg, T_DIM * NC);
        message_kernel<<<E_EDGES / 8, 256, 0, stream>>>(h, Wedge, ei, agg);
        combine_kernel<<<(T_DIM * N_NODES) / 8, 256, 0, stream>>>(h, agg, cnt, root_w, root_b);
    }

    out_kernel<<<(T_DIM * N_NODES) / 256, 256, 0, stream>>>(h, inv_w, inv_b, out);
}
